// XLSTM_Attention_Model_61091614819046
// MI455X (gfx1250) — matrix operand order repaired and checked by reading, not compiled, not run on hardware
//
#include <hip/hip_runtime.h>
#include <hip/hip_bf16.h>

// ---------------------------------------------------------------------------
// xLSTM (2-layer, 5-gate, exponential gating) + attention head for MI455X.
//   B=64, S=512, IN=64, H=512, gates G=5H=2560.
// Round 2: both W0 and W1 register-resident (400 VGPRs/wave of weights held
// across all 512 timesteps); A-fragment LDS loads software-pipelined so the
// wmma stream is not serialized on full s_wait_dscnt 0x0.
// ---------------------------------------------------------------------------

typedef __attribute__((ext_vector_type(16))) __bf16 v16bf;
typedef __attribute__((ext_vector_type(8)))  __bf16 v8bf;
typedef __attribute__((ext_vector_type(8)))  float  v8f;

constexpr int kB  = 64;
constexpr int kS  = 512;
constexpr int kIn = 64;
constexpr int kH  = 512;
constexpr int kG  = 5 * kH;          // 2560 fused gate columns
constexpr int K0  = kIn + kH;        // 576  (layer0 reduction dim)
constexpr int K1  = 2 * kH;          // 1024 (layer1 reduction dim)
constexpr int KT0 = K0 / 32;         // 18 k-tiles (bf16 wmma K=32)
constexpr int KT1 = K1 / 32;         // 32 k-tiles
constexpr int NSL = 16;              // column-slice workgroups per batch group
constexpr int WAVES = 10;            // 160 gate cols per WG = 10 N-tiles of 16
constexpr int THREADS = WAVES * 32;  // 320 threads (wave32)

// ---- A-matrix fragment load (16x32 bf16, ISA 7.12.2 layout) ----------------
// lane m = lane&15 selects the M row; lane>>4 selects 8-element K groups.
// Each 16x16 half is 16 contiguous bytes -> two aligned ds_load_b128 per lane.
__device__ __forceinline__ v16bf load_a_frag(const __bf16* base, int stride,
                                             int lane, int kcol) {
  int m = lane & 15, sel = lane >> 4;
  const __bf16* p = base + m * stride + kcol + sel * 8;
  v8bf lo = *(const v8bf*)(p);
  v8bf hi = *(const v8bf*)(p + 16);
  return __builtin_shufflevector(lo, hi, 0,1,2,3,4,5,6,7,8,9,10,11,12,13,14,15);
}

__device__ __forceinline__ void cluster_sync_release_acquire() {
  __builtin_amdgcn_fence(__ATOMIC_RELEASE, "agent");
#if __has_builtin(__builtin_amdgcn_s_cluster_barrier)
  __builtin_amdgcn_s_cluster_barrier();
#else
  asm volatile("s_barrier_signal -3\n\ts_barrier_wait -3" ::: "memory");
#endif
  __builtin_amdgcn_fence(__ATOMIC_ACQUIRE, "agent");
}

__device__ __forceinline__ float sigmoidf_fast(float v) {
  return 1.0f / (1.0f + __expf(-v));
}

// ---- weight pre-pack: fp32 [K][2560] -> bf16 B-fragments -------------------
// dst layout: [ns][wave][kt][lane][elem(16)], each lane's v16bf contiguous/32B.
// B fragment (32x16, ISA layout): n = lane&15 ; K = (j&8)*2 + (lane>>4)*8 + (j&7).
__global__ void pack_w_kernel(const float* __restrict__ W, __bf16* __restrict__ dst,
                              int KT) {
  size_t d = (size_t)blockIdx.x * blockDim.x + threadIdx.x;
  size_t total = (size_t)NSL * WAVES * KT * 512;
  if (d >= total) return;
  int j    = (int)(d & 15);
  int lane = (int)((d >> 4) & 31);
  size_t r = d >> 9;
  int kt = (int)(r % KT); r /= KT;
  int wv = (int)(r % WAVES);
  int ns = (int)(r / WAVES);
  int k   = kt * 32 + (j & 8) * 2 + (lane >> 4) * 8 + (j & 7);
  int col = (wv >> 1) * kH + ns * 32 + (wv & 1) * 16 + (lane & 15);
  dst[d] = (__bf16)W[(size_t)k * kG + col];
}

// ---- main recurrent kernel -------------------------------------------------
// grid = 64 WGs: bg = blockIdx/16 (16 batch rows), ns = blockIdx%16 (32 H cols).
// 2 cluster barriers per step separate read-h(t-1) / write-h(t) phases.
__global__ void __launch_bounds__(THREADS, 1)
xlstm_main_kernel(const float* __restrict__ x,
                  const float* __restrict__ b0,
                  const float* __restrict__ b1,
                  const __bf16* __restrict__ wsW0,
                  const __bf16* __restrict__ wsW1,
                  __bf16* __restrict__ hg0,       // [64][512] current h0
                  __bf16* __restrict__ lstm_out)  // [64][512][512] h1 history
{
  constexpr int A0S = K0 + 16;   // 592  (pad keeps 16B alignment, breaks conflicts)
  constexpr int A1S = K1 + 16;   // 1040
  __shared__ __bf16 a0[16 * A0S];
  __shared__ __bf16 a1[16 * A1S];
  __shared__ float  gts[WAVES * 256];   // per-wave 16x16 gate tiles
  __shared__ float  c0s[512], c1s[512]; // cell state slices [16 rows][32 cols]

  const int tid  = threadIdx.x;
  const int lane = tid & 31;
  const int wv   = tid >> 5;
  const int bg   = blockIdx.x / NSL;
  const int ns   = blockIdx.x % NSL;
  const int rowg = bg * 16;

  for (int e = tid; e < 512; e += THREADS) { c0s[e] = 0.f; c1s[e] = 0.f; }

  // Weights register-resident for the entire recurrence:
  //   layer0: 18 k-tiles x 8 VGPRs = 144 ; layer1: 32 x 8 = 256 ; total 400.
  v16bf w0f[KT0];
  v16bf w1f[KT1];
  {
    const __bf16* base0 = wsW0 + ((size_t)(ns * WAVES + wv) * KT0) * 512 + lane * 16;
#pragma unroll
    for (int kt = 0; kt < KT0; ++kt)
      w0f[kt] = *(const v16bf*)(base0 + (size_t)kt * 512);
    const __bf16* base1 = wsW1 + ((size_t)(ns * WAVES + wv) * KT1) * 512 + lane * 16;
#pragma unroll
    for (int kt = 0; kt < KT1; ++kt)
      w1f[kt] = *(const v16bf*)(base1 + (size_t)kt * 512);
  }

  __syncthreads();
  cluster_sync_release_acquire();   // hg0 was zeroed host-side; all WGs aligned

  for (int t = 0; t < kS; ++t) {
    // -- stage A0 = [x_t | h0(t-1)] into LDS (reads hg0 BEFORE barrier G1) --
    for (int idx = tid; idx < 16 * K0; idx += THREADS) {
      int m = idx / K0, k = idx - m * K0;
      __bf16 v = (k < kIn)
          ? (__bf16)x[((size_t)(rowg + m) * kS + t) * kIn + k]
          : hg0[(size_t)(rowg + m) * kH + (k - kIn)];
      a0[m * A0S + k] = v;
    }
    if (t + 1 < kS && tid < 16)     // warm L2/WGP$ for next step's x rows
      __builtin_prefetch(&x[((size_t)(rowg + tid) * kS + (t + 1)) * kIn], 0, 0);
    __syncthreads();
    cluster_sync_release_acquire(); // G1: everyone has consumed h0(t-1)

    // -- layer0 GEMM: 18 wmma, A loads pipelined 1 k-tile ahead --
    v8f acc0 = {};
    {
      v16bf a_cur = load_a_frag(a0, A0S, lane, 0);
#pragma unroll
      for (int kt = 0; kt < KT0; ++kt) {
        v16bf a_nxt = a_cur;
        if (kt + 1 < KT0) a_nxt = load_a_frag(a0, A0S, lane, (kt + 1) * 32);
        acc0 = __builtin_amdgcn_wmma_f32_16x16x32_bf16(false, a_cur, false, w0f[kt],
                                                       (short)0, acc0, false, false);
        a_cur = a_nxt;
      }
    }
    {
      int n = lane & 15, mh = (lane >> 4) * 8;
#pragma unroll
      for (int r = 0; r < 8; ++r) gts[wv * 256 + (mh + r) * 16 + n] = acc0[r];
    }
    __syncthreads();

    // -- layer0 elementwise on the 16x32 owned slice --
    for (int e = tid; e < 512; e += THREADS) {
      int m = e >> 5, n = e & 31, hf = n >> 4, nn = n & 15;
      int col = ns * 32 + n;
      float f  = gts[(0 * 2 + hf) * 256 + m * 16 + nn] + b0[0 * kH + col];
      float i  = gts[(1 * 2 + hf) * 256 + m * 16 + nn] + b0[1 * kH + col];
      float ch = gts[(2 * 2 + hf) * 256 + m * 16 + nn] + b0[2 * kH + col];
      float o  = gts[(3 * 2 + hf) * 256 + m * 16 + nn] + b0[3 * kH + col];
      float eg = gts[(4 * 2 + hf) * 256 + m * 16 + nn] + b0[4 * kH + col];
      f = sigmoidf_fast(f); i = sigmoidf_fast(i);
      o = sigmoidf_fast(o); eg = sigmoidf_fast(eg);
      ch = tanhf(ch);
      float c = f * c0s[e] + i * ch; c0s[e] = c;
      float hl = o * tanhf(c);
      float h  = eg * __expf(hl) + (1.0f - eg) * hl;
      hg0[(size_t)(rowg + m) * kH + col] = (__bf16)h;
    }
    __syncthreads();
    cluster_sync_release_acquire(); // G2: h0(t) visible to all slice WGs

    // -- stage A1 = [h0(t) | h1(t-1)] --
    for (int idx = tid; idx < 16 * K1; idx += THREADS) {
      int m = idx >> 10, k = idx & 1023;
      __bf16 v;
      if (k < kH)      v = hg0[(size_t)(rowg + m) * kH + k];
      else if (t == 0) v = (__bf16)0.0f;
      else             v = lstm_out[((size_t)(rowg + m) * kS + (t - 1)) * kH + (k - kH)];
      a1[m * A1S + k] = v;
    }
    __syncthreads();

    // -- layer1 GEMM: 32 wmma, register-resident W1, pipelined A loads --
    v8f acc1 = {};
    {
      v16bf a_cur = load_a_frag(a1, A1S, lane, 0);
#pragma unroll
      for (int kt = 0; kt < KT1; ++kt) {
        v16bf a_nxt = a_cur;
        if (kt + 1 < KT1) a_nxt = load_a_frag(a1, A1S, lane, (kt + 1) * 32);
        acc1 = __builtin_amdgcn_wmma_f32_16x16x32_bf16(false, a_cur, false, w1f[kt],
                                                       (short)0, acc1, false, false);
        a_cur = a_nxt;
      }
    }
    {
      int n = lane & 15, mh = (lane >> 4) * 8;
#pragma unroll
      for (int r = 0; r < 8; ++r) gts[wv * 256 + (mh + r) * 16 + n] = acc1[r];
    }
    __syncthreads();

    // -- layer1 elementwise; h1(t) slice -> lstm_out (disjoint t rows: no race)
    for (int e = tid; e < 512; e += THREADS) {
      int m = e >> 5, n = e & 31, hf = n >> 4, nn = n & 15;
      int col = ns * 32 + n;
      float f  = gts[(0 * 2 + hf) * 256 + m * 16 + nn] + b1[0 * kH + col];
      float i  = gts[(1 * 2 + hf) * 256 + m * 16 + nn] + b1[1 * kH + col];
      float ch = gts[(2 * 2 + hf) * 256 + m * 16 + nn] + b1[2 * kH + col];
      float o  = gts[(3 * 2 + hf) * 256 + m * 16 + nn] + b1[3 * kH + col];
      float eg = gts[(4 * 2 + hf) * 256 + m * 16 + nn] + b1[4 * kH + col];
      f = sigmoidf_fast(f); i = sigmoidf_fast(i);
      o = sigmoidf_fast(o); eg = sigmoidf_fast(eg);
      ch = tanhf(ch);
      float c = f * c1s[e] + i * ch; c1s[e] = c;
      float hl = o * tanhf(c);
      float h  = eg * __expf(hl) + (1.0f - eg) * hl;
      lstm_out[((size_t)(rowg + m) * kS + t) * kH + col] = (__bf16)h;
    }
    // next iteration's G1/G2 order the cross-WG reads of lstm_out(t) / hg0(t)
  }
}

// ---- attention + fc head ---------------------------------------------------
__global__ void __launch_bounds__(256, 1)
attn_kernel(const __bf16* __restrict__ lstm_out,
            const float* __restrict__ Wa,  const float* __restrict__ ba,
            const float* __restrict__ Wfc, const float* __restrict__ bfc,
            float* __restrict__ out)  // [0..63]=out, [64..]=att flat
{
  __shared__ float lg[kS];
  __shared__ float red[256];
  const int b = blockIdx.x, tid = threadIdx.x;
  const __bf16* base = lstm_out + (size_t)b * kS * kH;

  for (int t = tid; t < kS; t += 256) {
    const __bf16* row = base + (size_t)t * kH;
    float s = 0.f;
    for (int j = 0; j < kH; ++j) s += (float)row[j] * Wa[j];
    lg[t] = s + ba[0];
  }
  __syncthreads();

  float m = -3.0e30f;
  for (int t = tid; t < kS; t += 256) m = fmaxf(m, lg[t]);
  red[tid] = m; __syncthreads();
  for (int s2 = 128; s2 > 0; s2 >>= 1) {
    if (tid < s2) red[tid] = fmaxf(red[tid], red[tid + s2]);
    __syncthreads();
  }
  float mx = red[0]; __syncthreads();

  float ps = 0.f;
  for (int t = tid; t < kS; t += 256) { float e = __expf(lg[t] - mx); lg[t] = e; ps += e; }
  red[tid] = ps; __syncthreads();
  for (int s2 = 128; s2 > 0; s2 >>= 1) {
    if (tid < s2) red[tid] += red[tid + s2];
    __syncthreads();
  }
  float inv = 1.0f / red[0]; __syncthreads();

  for (int t = tid; t < kS; t += 256) { lg[t] *= inv; out[kB + b * kS + t] = lg[t]; }
  __syncthreads();

  float po = 0.f;
  for (int j = tid; j < kH; j += 256) {
    float s = 0.f;
    for (int t = 0; t < kS; ++t) s += lg[t] * (float)base[(size_t)t * kH + j];
    po += s * Wfc[j];
  }
  red[tid] = po; __syncthreads();
  for (int s2 = 128; s2 > 0; s2 >>= 1) {
    if (tid < s2) red[tid] += red[tid + s2];
    __syncthreads();
  }
  if (tid == 0) out[b] = red[0] + bfc[0];
}

// ---------------------------------------------------------------------------
extern "C" void kernel_launch(void* const* d_in, const int* in_sizes, int n_in,
                              void* d_out, int out_size, void* d_ws, size_t ws_size,
                              hipStream_t stream) {
  (void)in_sizes; (void)n_in; (void)out_size; (void)ws_size;
  const float* x   = (const float*)d_in[0];
  const float* W0  = (const float*)d_in[1];
  const float* b0  = (const float*)d_in[2];
  const float* W1  = (const float*)d_in[3];
  const float* b1  = (const float*)d_in[4];
  const float* Wa  = (const float*)d_in[5];
  const float* ba  = (const float*)d_in[6];
  const float* Wfc = (const float*)d_in[7];
  const float* bfc = (const float*)d_in[8];
  float* out = (float*)d_out;

  // workspace carve-out (256B aligned): packed W0, packed W1, hg0, lstm_out
  const size_t W0E = (size_t)NSL * WAVES * KT0 * 512;  // 1,474,560 bf16
  const size_t W1E = (size_t)NSL * WAVES * KT1 * 512;  // 2,621,440 bf16
  const size_t HGE = (size_t)kB * kH;                  // 32,768 bf16
  const size_t LOE = (size_t)kB * kS * kH;             // 16,777,216 bf16
  char* ws = (char*)d_ws;
  size_t off = 0;
  auto carve = [&](size_t bytes) {
    size_t p = off;
    off = (off + bytes + 255) & ~(size_t)255;
    return p;
  };
  __bf16* wsW0 = (__bf16*)(ws + carve(W0E * sizeof(__bf16)));
  __bf16* wsW1 = (__bf16*)(ws + carve(W1E * sizeof(__bf16)));
  __bf16* hg0  = (__bf16*)(ws + carve(HGE * sizeof(__bf16)));
  __bf16* lstm = (__bf16*)(ws + carve(LOE * sizeof(__bf16)));

  hipMemsetAsync(hg0, 0, HGE * sizeof(__bf16), stream);  // h0(-1) = 0

  pack_w_kernel<<<(unsigned)((W0E + 255) / 256), 256, 0, stream>>>(W0, wsW0, KT0);
  pack_w_kernel<<<(unsigned)((W1E + 255) / 256), 256, 0, stream>>>(W1, wsW1, KT1);

  // NOTE: dispatch should configure 16-WG clusters (one per batch group) so the
  // s_cluster_barrier pairs synchronize the column-slice WGs each timestep.
  xlstm_main_kernel<<<dim3(kB / 16 * NSL), dim3(THREADS), 0, stream>>>(
      x, b0, b1, wsW0, wsW1, hg0, lstm);

  attn_kernel<<<dim3(kB), dim3(256), 0, stream>>>(lstm, Wa, ba, Wfc, bfc, out);
}
